// seq2seq_42391327211814
// MI455X (gfx1250) — compile-verified, matrix operand order checked by reading, not run on hardware
//
#include <hip/hip_runtime.h>
#include <math.h>

#define B_SZ 64
#define T_SZ 512
#define H_SZ 1024
#define HD_SZ 2048

typedef __attribute__((ext_vector_type(16))) __bf16 v16bf;
typedef __attribute__((ext_vector_type(8)))  float  v8f;
typedef __attribute__((ext_vector_type(4))) unsigned int u32x4;
typedef __attribute__((ext_vector_type(8))) int         i32x8;
typedef __attribute__((ext_vector_type(4))) int         i32x4;

#if __has_builtin(__builtin_amdgcn_tensor_load_to_lds) && \
    __has_builtin(__builtin_amdgcn_s_wait_tensorcnt)
#define HAVE_TDM 1
#else
#define HAVE_TDM 0
#endif

// ---------------------------------------------------------------------------
// small device helpers
// ---------------------------------------------------------------------------
__device__ __forceinline__ unsigned short f2bf(float f) {
  unsigned int u = __float_as_uint(f);
  unsigned int r = (u + 0x7FFFu + ((u >> 16) & 1u)) >> 16;  // RNE
  return (unsigned short)r;
}

__device__ __forceinline__ __bf16 bfbits(unsigned short h) {
  return __builtin_bit_cast(__bf16, h);
}

__device__ __forceinline__ v8f v8f_zero() {
  v8f z;
#pragma unroll
  for (int i = 0; i < 8; ++i) z[i] = 0.0f;
  return z;
}

__device__ __forceinline__ v8f wmma_bf16(v16bf a, v16bf b, v8f c) {
  // (neg_a, A, neg_b, B, c_mod, C, reuse_a, reuse_b)
  return __builtin_amdgcn_wmma_f32_16x16x32_bf16(false, a, false, b, (short)0,
                                                 c, false, false);
}

// Load a 16x32 bf16 operand fragment per the CDNA5 VGPR layout (ISA 7.12.2).
// A: src -> [row0, k0] of row-major [M,K]; lane lo = row.
// B: pass W row-major [N,K] (gh = h @ W^T); lane lo = output column n.
__device__ __forceinline__ v16bf load_frag(const unsigned short* __restrict__ src,
                                           int ld) {
  const int lane = threadIdx.x & 31;
  const int lo = lane & 15;
  const int hi = lane >> 4;
  v16bf f;
#pragma unroll
  for (int v = 0; v < 8; ++v) {
    const int kk = ((v & 4) << 2) + hi * 8 + (v & 3) * 2;  // K index of elem 2v
    const unsigned short* p = src + lo * ld + kk;
    f[2 * v]     = bfbits(p[0]);
    f[2 * v + 1] = bfbits(p[1]);
  }
  return f;
}

__device__ __forceinline__ float sigmoidf_(float x) {
  return 1.0f / (1.0f + __expf(-x));
}

// ---------------------------------------------------------------------------
// utility kernels
// ---------------------------------------------------------------------------
__global__ void f32_to_bf16_kernel(const float* __restrict__ s,
                                   unsigned short* __restrict__ d, int n) {
  int i = blockIdx.x * blockDim.x + threadIdx.x;
  if (i < n) d[i] = f2bf(s[i]);
}

// input_tensor [B, T, 64] fp32  ->  X0 [T, B, 64] bf16
__global__ void transpose_x_kernel(const float* __restrict__ x,
                                   unsigned short* __restrict__ x0) {
  int i = blockIdx.x * blockDim.x + threadIdx.x;
  if (i >= B_SZ * T_SZ * 64) return;
  int d = i & 63;
  int t = (i >> 6) & (T_SZ - 1);
  int b = i >> 15;
  x0[((size_t)(t * B_SZ + b)) * 64 + d] = f2bf(x[i]);
}

__global__ void zero32_kernel(unsigned int* __restrict__ p, int n) {
  int i = blockIdx.x * blockDim.x + threadIdx.x;
  if (i < n) p[i] = 0u;
}

// Final hidden of last encoder layer (fp32 [64,1024]) -> enc_hidden half in
// d_out, and decoder initial hidden (fp32 + bf16).
__global__ void copy_final_kernel(const float* __restrict__ hsrc,
                                  float* __restrict__ out_enc,
                                  float* __restrict__ dec_hf,
                                  unsigned short* __restrict__ dec_hb,
                                  int half) {
  int i = blockIdx.x * blockDim.x + threadIdx.x;
  if (i >= B_SZ * H_SZ) return;
  int b = i >> 10;
  int j = i & (H_SZ - 1);
  float v = hsrc[i];
  int o = b * HD_SZ + half * H_SZ + j;
  out_enc[o] = v;
  dec_hf[o] = v;
  dec_hb[o] = f2bf(v);
}

// ---------------------------------------------------------------------------
// batched input-gate GEMM:  Gi[m,n] = sum_k X[m,k]*Wi[n,k] + bi[n]
// X: [M, KIN] bf16 row-major, Wi: [3072, KIN] bf16 row-major, Gi fp32 [M,3072]
// one wave = 16 rows x 64 cols
// ---------------------------------------------------------------------------
template <int KIN>
__global__ __launch_bounds__(128) void gi_gemm_kernel(
    const unsigned short* __restrict__ X, const unsigned short* __restrict__ Wi,
    const float* __restrict__ bi, float* __restrict__ Gi, int M) {
  const int tid = threadIdx.x;
  const int gw = blockIdx.x * (blockDim.x >> 5) + (tid >> 5);
  const int mtiles = M >> 4;
  const int mt = gw % mtiles;
  const int nc = gw / mtiles;  // 64-col chunk, 48 total
  const int lane = tid & 31;
  const int lo = lane & 15;
  const int hi = lane >> 4;

  v8f acc[4];
#pragma unroll
  for (int n = 0; n < 4; ++n) acc[n] = v8f_zero();

  const unsigned short* Arow = X + (size_t)mt * 16 * KIN;
#pragma unroll 1
  for (int k = 0; k < KIN; k += 32) {
    v16bf a = load_frag(Arow + k, KIN);
    v16bf bb[4];
#pragma unroll
    for (int n = 0; n < 4; ++n)
      bb[n] = load_frag(Wi + (size_t)(nc * 64 + n * 16) * KIN + k, KIN);
#pragma unroll
    for (int n = 0; n < 4; ++n) acc[n] = wmma_bf16(a, bb[n], acc[n]);
  }

#pragma unroll
  for (int n = 0; n < 4; ++n) {
#pragma unroll
    for (int v = 0; v < 8; ++v) {
      int m = mt * 16 + v + hi * 8;
      int j = nc * 64 + n * 16 + lo;
      Gi[(size_t)m * 3072 + j] = acc[n][v] + bi[j];
    }
  }
}

// ---------------------------------------------------------------------------
// one GRU timestep (recurrent part + gates):
//   gh = h_prev @ Wh^T ;  r,z = sigmoid(gi+gh+bh) ; n = tanh(gi_n + r*(gh_n+bh_n))
//   h = mask ? (1-z)*n + z*h_prev : h_prev
// K = hidden size (1024 encoder / 2048 decoder); gate width == K.
// grid: K/64 blocks of 128 threads; wave w owns columns [blk*64+w*16, +16)
// and all three gates for them (12 accumulators).
// The full 64xK bf16 h_prev tile (128/256 KB, fits the 320 KB WGP LDS) is
// DMA'd into LDS with ONE Tensor-Data-Mover descriptor issued by wave 0,
// synchronized via TENSORcnt + one barrier; the WMMA k-loop then reads A
// fragments from LDS with no further barriers.
// ---------------------------------------------------------------------------
template <int K, bool GI_BCAST>
__global__ __launch_bounds__(128) void gru_step_kernel(
    const unsigned short* __restrict__ h_bf,  // [64, K] prev (bf16)
    const float* __restrict__ h_f,            // [64, K] prev (fp32)
    float* __restrict__ h_f_out,              // [64, K] next (fp32)
    unsigned short* __restrict__ h_bf_out,    // [64, K] next (bf16)
    const unsigned short* __restrict__ Wh,    // [3K, K] bf16
    const float* __restrict__ bh,             // [3K]
    const float* __restrict__ gi,  // enc: [64,3K] (this t, incl. bi); dec: [3K]
    const int* __restrict__ seq_len,          // null => no mask
    int t,
    unsigned short* __restrict__ y,           // null => skip; row (t*64+m)
    int ldy, int ycol) {
  extern __shared__ unsigned short Alds[];  // [64 * K] bf16 (dynamic LDS)

  const int tid = threadIdx.x;
  const int wave = tid >> 5;
  const int lane = tid & 31;
  const int lo = lane & 15;
  const int hi = lane >> 4;
  const int jbase = blockIdx.x * 64 + wave * 16;

#if HAVE_TDM
  if (wave == 0) {
    unsigned long long ga = (unsigned long long)(uintptr_t)h_bf;
    unsigned lba = (unsigned)(uintptr_t)Alds;  // LDS byte offset (addr[31:0])
    // D# group 0: count=1 | lds_addr | global_addr[56:0] | type=2
    u32x4 g0;
    g0[0] = 1u;
    g0[1] = lba;
    g0[2] = (unsigned)(ga & 0xFFFFFFFFu);
    g0[3] = (unsigned)((ga >> 32) & 0x01FFFFFFu) | (2u << 30);
    // D# group 1: wg_mask=0, data_size=2B, tensor 64xK == tile 64xK, stride K
    i32x8 g1;
    g1[0] = (int)(1u << 16);                              // data_size = 2 bytes
    g1[1] = (int)(((unsigned)K & 0xFFFFu) << 16);         // tensor_dim0 lo16
    g1[2] = (int)(((unsigned)K >> 16) | (64u << 16));     // dim0 hi | dim1 lo
    g1[3] = (int)((unsigned)K << 16);                     // dim1 hi=0 | tile_dim0
    g1[4] = 64;                                           // tile_dim1 | tile_dim2=0
    g1[5] = K;                                            // tensor_dim0_stride lo
    g1[6] = 0;
    g1[7] = 0;
    i32x4 z4 = {0, 0, 0, 0};
#if __clang_major__ >= 23
    i32x8 z8 = {0, 0, 0, 0, 0, 0, 0, 0};
    __builtin_amdgcn_tensor_load_to_lds(g0, g1, z4, z4, z8, 0);
#else
    __builtin_amdgcn_tensor_load_to_lds(g0, g1, z4, z4, 0);
#endif
    __builtin_amdgcn_s_wait_tensorcnt(0);
  }
#else
  // fallback: cooperative copy h_prev -> LDS
  for (int i = tid; i < 64 * K / 2; i += 128) {
    ((unsigned int*)Alds)[i] = ((const unsigned int*)h_bf)[i];
  }
#endif
  __syncthreads();

  v8f acc[3][4];
#pragma unroll
  for (int g = 0; g < 3; ++g)
#pragma unroll
    for (int m = 0; m < 4; ++m) acc[g][m] = v8f_zero();

#pragma unroll 1
  for (int k = 0; k < K; k += 32) {
    v16bf a0 = load_frag(&Alds[0 * 16 * K + k], K);
    v16bf a1 = load_frag(&Alds[1 * 16 * K + k], K);
    v16bf a2 = load_frag(&Alds[2 * 16 * K + k], K);
    v16bf a3 = load_frag(&Alds[3 * 16 * K + k], K);
    v16bf bb[3];
#pragma unroll
    for (int g = 0; g < 3; ++g)
      bb[g] = load_frag(Wh + (size_t)(g * K + jbase) * K + k, K);
#pragma unroll
    for (int g = 0; g < 3; ++g) {
      acc[g][0] = wmma_bf16(a0, bb[g], acc[g][0]);
      acc[g][1] = wmma_bf16(a1, bb[g], acc[g][1]);
      acc[g][2] = wmma_bf16(a2, bb[g], acc[g][2]);
      acc[g][3] = wmma_bf16(a3, bb[g], acc[g][3]);
    }
  }

  const int j = jbase + lo;
  const float bhr = bh[j];
  const float bhz = bh[K + j];
  const float bhn = bh[2 * K + j];

#pragma unroll
  for (int mt = 0; mt < 4; ++mt) {
#pragma unroll
    for (int v = 0; v < 8; ++v) {
      int m = mt * 16 + v + hi * 8;
      float gir, giz, gin;
      if (GI_BCAST) {
        gir = gi[j];
        giz = gi[K + j];
        gin = gi[2 * K + j];
      } else {
        const float* grow = gi + (size_t)m * (3 * K);
        gir = grow[j];
        giz = grow[K + j];
        gin = grow[2 * K + j];
      }
      float r = sigmoidf_(gir + acc[0][mt][v] + bhr);
      float z = sigmoidf_(giz + acc[1][mt][v] + bhz);
      float n = tanhf(gin + r * (acc[2][mt][v] + bhn));
      float hold = h_f[(size_t)m * K + j];
      float hnew = (1.0f - z) * n + z * hold;
      bool valid = seq_len ? (t < seq_len[m]) : true;
      float ho = valid ? hnew : hold;
      h_f_out[(size_t)m * K + j] = ho;
      unsigned short hb = f2bf(ho);
      h_bf_out[(size_t)m * K + j] = hb;
      if (y) y[(size_t)(t * B_SZ + m) * ldy + ycol + j] = hb;
    }
  }
}

// ---------------------------------------------------------------------------
// decoder output projection: out[b,t,d] = sum_k decH[t*64+b,k]*Wo[d,k] + bo[d]
// decH: [32768, 2048] bf16 ; Wo: [64, 2048] bf16 ; one wave = 16 rows x 64 cols
// ---------------------------------------------------------------------------
__global__ __launch_bounds__(128) void proj_gemm_kernel(
    const unsigned short* __restrict__ A, const unsigned short* __restrict__ W,
    const float* __restrict__ bo, float* __restrict__ out) {
  const int tid = threadIdx.x;
  const int gw = blockIdx.x * 4 + (tid >> 5);  // m-tile id, 2048 total
  const int lane = tid & 31;
  const int lo = lane & 15;
  const int hi = lane >> 4;

  v8f acc[4];
#pragma unroll
  for (int n = 0; n < 4; ++n) acc[n] = v8f_zero();

  const unsigned short* Arow = A + (size_t)gw * 16 * HD_SZ;
#pragma unroll 1
  for (int k = 0; k < HD_SZ; k += 32) {
    v16bf a = load_frag(Arow + k, HD_SZ);
    v16bf bb[4];
#pragma unroll
    for (int n = 0; n < 4; ++n)
      bb[n] = load_frag(W + (size_t)(n * 16) * HD_SZ + k, HD_SZ);
#pragma unroll
    for (int n = 0; n < 4; ++n) acc[n] = wmma_bf16(a, bb[n], acc[n]);
  }

#pragma unroll
  for (int n = 0; n < 4; ++n) {
#pragma unroll
    for (int v = 0; v < 8; ++v) {
      int gm = gw * 16 + v + hi * 8;  // = t*64 + b
      int t = gm >> 6;
      int b = gm & 63;
      int d = n * 16 + lo;
      out[((size_t)b * T_SZ + t) * 64 + d] = acc[n][v] + bo[d];
    }
  }
}

// ---------------------------------------------------------------------------
// host
// ---------------------------------------------------------------------------
extern "C" void kernel_launch(void* const* d_in, const int* in_sizes, int n_in,
                              void* d_out, int out_size, void* d_ws,
                              size_t ws_size, hipStream_t stream) {
  (void)in_sizes; (void)n_in; (void)out_size; (void)ws_size;

  const float* x_in = (const float*)d_in[0];
  const int* seq_len = (const int*)d_in[1];
  const float *eWi[6], *eWh[6], *ebi[6], *ebh[6];
  for (int i = 0; i < 6; ++i) {  // i = layer*2 + dir (0=fwd)
    int base = 2 + i * 4;
    eWi[i] = (const float*)d_in[base + 0];
    eWh[i] = (const float*)d_in[base + 1];
    ebi[i] = (const float*)d_in[base + 2];
    ebh[i] = (const float*)d_in[base + 3];
  }
  const float* dWh_f = (const float*)d_in[27];
  const float* dbi   = (const float*)d_in[28];
  const float* dbh   = (const float*)d_in[29];
  const float* Wo_f  = (const float*)d_in[30];
  const float* bo    = (const float*)d_in[31];
  float* out = (float*)d_out;  // [64*2048 enc_hidden][64*512*64 decoder_output]

  // ---- workspace carve-up ----
  char* W = (char*)d_ws;
  size_t off = 0;
  auto take = [&](size_t bytes) -> void* {
    void* p = W + off;
    off += (bytes + 255) & ~(size_t)255;
    return p;
  };
  unsigned short *Wi_b[6], *Wh_b[6];
  for (int i = 0; i < 6; ++i) {
    size_t kin = (i < 2) ? 64 : 2048;
    Wi_b[i] = (unsigned short*)take((size_t)3072 * kin * 2);
    Wh_b[i] = (unsigned short*)take((size_t)3072 * 1024 * 2);
  }
  unsigned short* dWh_b = (unsigned short*)take((size_t)6144 * 2048 * 2);
  unsigned short* Wo_b  = (unsigned short*)take((size_t)64 * 2048 * 2);
  unsigned short* X0    = (unsigned short*)take((size_t)T_SZ * B_SZ * 64 * 2);
  unsigned short* bufA  = (unsigned short*)take((size_t)T_SZ * B_SZ * 2048 * 2);
  unsigned short* bufB  = (unsigned short*)take((size_t)T_SZ * B_SZ * 2048 * 2);
  float* Gi             = (float*)take((size_t)T_SZ * B_SZ * 3072 * 4);
  float* hF[2];          unsigned short* hB[2];
  float* dF[2];          unsigned short* dB[2];
  for (int i = 0; i < 2; ++i) {
    hF[i] = (float*)take((size_t)B_SZ * HD_SZ * 4);
    hB[i] = (unsigned short*)take((size_t)B_SZ * HD_SZ * 2);
    dF[i] = (float*)take((size_t)B_SZ * HD_SZ * 4);
    dB[i] = (unsigned short*)take((size_t)B_SZ * HD_SZ * 2);
  }
  unsigned short* decH = bufA;  // reused after encoder finishes with it

  auto conv = [&](const float* s, unsigned short* d, size_t n) {
    f32_to_bf16_kernel<<<(unsigned)((n + 255) / 256), 256, 0, stream>>>(s, d, (int)n);
  };
  auto zero = [&](void* p, size_t n_u32) {
    zero32_kernel<<<(unsigned)((n_u32 + 255) / 256), 256, 0, stream>>>((unsigned int*)p, (int)n_u32);
  };

  // ---- weight conversion to bf16 (all stay L2-resident, ~113 MB) ----
  for (int i = 0; i < 6; ++i) {
    size_t kin = (i < 2) ? 64 : 2048;
    conv(eWi[i], Wi_b[i], (size_t)3072 * kin);
    conv(eWh[i], Wh_b[i], (size_t)3072 * 1024);
  }
  conv(dWh_f, dWh_b, (size_t)6144 * 2048);
  conv(Wo_f, Wo_b, (size_t)64 * 2048);
  transpose_x_kernel<<<(B_SZ * T_SZ * 64 + 255) / 256, 256, 0, stream>>>(x_in, X0);

  const int M = T_SZ * B_SZ;  // 32768
  const size_t lds_enc = (size_t)64 * H_SZ * 2;   // 128 KB
  const size_t lds_dec = (size_t)64 * HD_SZ * 2;  // 256 KB (< 320 KB WGP LDS)

  // ---- encoder: for each layer, fwd scan then bwd scan; Gi buffer reused ----
  auto run_scan = [&](int widx, const unsigned short* Xl, int kin,
                      unsigned short* Yout, int ycol, int reverse) {
    // batched input-gate GEMM (off the serial path)
    int waves = (M / 16) * 48;
    if (kin == 64)
      gi_gemm_kernel<64><<<waves / 4, 128, 0, stream>>>(Xl, Wi_b[widx], ebi[widx], Gi, M);
    else
      gi_gemm_kernel<2048><<<waves / 4, 128, 0, stream>>>(Xl, Wi_b[widx], ebi[widx], Gi, M);
    // h0 = 0
    zero(hF[0], (size_t)B_SZ * H_SZ);
    zero(hB[0], (size_t)B_SZ * H_SZ / 2);
    // serial recurrence: 512 small WMMA step kernels, ping-pong state
    for (int s = 0; s < T_SZ; ++s) {
      int t = reverse ? (T_SZ - 1 - s) : s;
      int cur = s & 1, nxt = cur ^ 1;
      gru_step_kernel<H_SZ, false><<<H_SZ / 64, 128, lds_enc, stream>>>(
          hB[cur], hF[cur], hF[nxt], hB[nxt], Wh_b[widx], ebh[widx],
          Gi + (size_t)t * B_SZ * 3072, seq_len, t, Yout, 2048, ycol);
    }
    // 512 steps (even) -> final state back in index 0
  };

  // layer 0: X0 [T,B,64] -> bufA
  run_scan(0, X0, 64, bufA, 0, 0);
  run_scan(1, X0, 64, bufA, H_SZ, 1);
  // layer 1: bufA -> bufB
  run_scan(2, bufA, 2048, bufB, 0, 0);
  run_scan(3, bufA, 2048, bufB, H_SZ, 1);
  // layer 2: bufB -> (y unused); final h -> enc_hidden + decoder init
  run_scan(4, bufB, 2048, nullptr, 0, 0);
  copy_final_kernel<<<(B_SZ * H_SZ + 255) / 256, 256, 0, stream>>>(hF[0], out, dF[0], dB[0], 0);
  run_scan(5, bufB, 2048, nullptr, 0, 1);
  copy_final_kernel<<<(B_SZ * H_SZ + 255) / 256, 256, 0, stream>>>(hF[0], out, dF[0], dB[0], 1);

  // ---- decoder: zero input => gi == dbi (broadcast); log h in bf16 ----
  for (int t = 0; t < T_SZ; ++t) {
    int cur = t & 1, nxt = cur ^ 1;
    gru_step_kernel<HD_SZ, true><<<HD_SZ / 64, 128, lds_dec, stream>>>(
        dB[cur], dF[cur], dF[nxt], dB[nxt], dWh_b, dbh, dbi,
        nullptr, t, decH, 2048, 0);
  }

  // ---- batched output projection straight into d_out ----
  proj_gemm_kernel<<<(M / 16) / 4, 128, 0, stream>>>(decH, Wo_b, bo,
                                                     out + (size_t)B_SZ * HD_SZ);
}
